// NodeModel_17317308137511
// MI455X (gfx1250) — matrix operand order repaired and checked by reading, not compiled, not run on hardware
//
#include <hip/hip_runtime.h>
#include <hip/hip_bf16.h>

typedef _Float16 f16;
typedef __attribute__((ext_vector_type(16))) _Float16 v16h;
typedef __attribute__((ext_vector_type(8)))  _Float16 v8h;
typedef __attribute__((ext_vector_type(8)))  float    v8f;

#define OUT_STATS   0
#define OUT_SCATTER 1
#define OUT_F32     2

__device__ __forceinline__ void gadd(float* p, float v) { unsafeAtomicAdd(p, v); }

// A operand (16x32 f16 tile slice for this lane): lo lanes hold K k0+0..7 (v0-3)
// and k0+16..23 (v4-7); hi lanes are shifted by +8. Caller passes p already
// offset by (k0 + hi*8); second half is at +16 elements.
__device__ __forceinline__ v16h ld_a_h(const f16* p) {
  v8h a0 = *(const v8h*)p;
  v8h a1 = *(const v8h*)(p + 16);
  v16h r;
#pragma unroll
  for (int i = 0; i < 8; ++i) { r[i] = a0[i]; r[i + 8] = a1[i]; }
  return r;
}
__device__ __forceinline__ v16h ld_a_f(const float* p) {
  v8f a0 = *(const v8f*)p;
  v8f a1 = *(const v8f*)(p + 16);
  v16h r;
#pragma unroll
  for (int i = 0; i < 8; ++i) { r[i] = (f16)a0[i]; r[i + 8] = (f16)a1[i]; }
  return r;
}

// ---------------------------------------------------------------------------
// Fused BN-folded GEMM:  out[r, 0:128] = act( A[r, 0:DIN] * Wf^T + cf )
// Block = 128 threads = 4 waves; each wave computes 32 rows x 128 cols via
// 2 M-tiles x 8 N-tiles of WMMA 16x16x32 (B fragments reused across M-tiles).
// Optional fused column stats (for next layer's BN) or fused scatter-add
// epilogue (segment sum).
// ---------------------------------------------------------------------------
template <int KCHUNKS, bool A16, bool GATHER, int W1, int OMODE, bool RELU>
__global__ __launch_bounds__(128) void k_gemm(
    const f16* __restrict__ hA,      // f16 A source (row width = KCHUNKS*32)
    const float* __restrict__ src0,  // f32 A cols 0..127 (x)
    const float* __restrict__ src1,  // f32 A cols 128.. (edge_attr / agg)
    const int* __restrict__ rowidx,  // gather index for src0 rows
    const int* __restrict__ colidx,  // scatter index (OUT_SCATTER)
    const f16* __restrict__ Wf,      // folded weights [128][DIN]
    const float* __restrict__ cf,    // folded bias [128]
    f16* __restrict__ hout,          // f16 activations out (OUT_STATS)
    float* __restrict__ fout,        // f32 out (OUT_F32)
    float* __restrict__ stats,       // [256]: sum[0:128], sumsq[128:256]
    float* __restrict__ agg,         // scatter target [N][128]
    int nrows) {
  constexpr int DIN = KCHUNKS * 32;
  __shared__ float ls[256];
  const int tid = threadIdx.x;
  if (OMODE == OUT_STATS) {
    ls[tid] = 0.f;
    ls[tid + 128] = 0.f;
    __syncthreads();
  }
  const int lane = tid & 31;
  const int wave = tid >> 5;
  const int hi = lane >> 4;    // 0 or 1
  const int lm = lane & 15;
  const long rowbase = (long)blockIdx.x * 128 + wave * 32;
  if (rowbase < nrows) {  // wave-uniform guard (nrows % 16 == 0)
    const bool valid1 = (rowbase + 16) < nrows;  // second M-tile validity
    v8f acc[2][8] = {};
    const long r0 = rowbase + lm;
    // clamp tile-1 A row so loads stay in-bounds; results discarded if !valid1
    const long r1 = valid1 ? (rowbase + 16 + lm) : r0;
#pragma unroll
    for (int kc = 0; kc < KCHUNKS; ++kc) {
      const int k0 = kc * 32;
      v16h a0, a1;
      if (A16) {
        a0 = ld_a_h(hA + r0 * DIN + k0 + hi * 8);
        a1 = ld_a_h(hA + r1 * DIN + k0 + hi * 8);
      } else {
        if (k0 < 128) {
          const long g0 = GATHER ? (long)rowidx[r0] : r0;
          const long g1 = GATHER ? (long)rowidx[r1] : r1;
          a0 = ld_a_f(src0 + g0 * 128 + k0 + hi * 8);
          a1 = ld_a_f(src0 + g1 * 128 + k0 + hi * 8);
        } else {
          a0 = ld_a_f(src1 + r0 * W1 + (k0 - 128) + hi * 8);
          a1 = ld_a_f(src1 + r1 * W1 + (k0 - 128) + hi * 8);
        }
      }
#pragma unroll
      for (int t = 0; t < 8; ++t) {
        // B layout: lane = column N (t*16+lm); lo lanes K k0+0..7 and k0+16..23, hi +8.
        const f16* bp = Wf + (size_t)(t * 16 + lm) * DIN + k0 + hi * 8;
        v8h b0 = *(const v8h*)bp;
        v8h b1 = *(const v8h*)(bp + 16);
        v16h b;
#pragma unroll
        for (int i = 0; i < 8; ++i) { b[i] = b0[i]; b[i + 8] = b1[i]; }
        acc[0][t] = __builtin_amdgcn_wmma_f32_16x16x32_f16(
            false, a0, false, b, (short)0, acc[0][t], false, false);
        acc[1][t] = __builtin_amdgcn_wmma_f32_16x16x32_f16(
            false, a1, false, b, (short)0, acc[1][t], false, false);
      }
    }
    // ---------------- epilogue (per M-tile, wave-uniform guard) -----------
#pragma unroll
    for (int mt = 0; mt < 2; ++mt) {
      if (mt == 1 && !valid1) break;
      const long mbase = rowbase + mt * 16;
      int cv[8];
      if (OMODE == OUT_SCATTER) {
#pragma unroll
        for (int j = 0; j < 8; ++j) cv[j] = colidx[mbase + hi * 8 + j];
      }
#pragma unroll
      for (int t = 0; t < 8; ++t) {
        const int colg = t * 16 + lm;
        const float bias = cf[colg];
        float s = 0.f, q = 0.f;
#pragma unroll
        for (int j = 0; j < 8; ++j) {  // C layout: VGPR j -> row M = j + hi*8
          float v = acc[mt][t][j] + bias;
          if (RELU) v = v > 0.f ? v : 0.f;
          acc[mt][t][j] = v;
          s += v;
          q += v * v;
        }
        if (OMODE == OUT_STATS) {
#pragma unroll
          for (int j = 0; j < 8; ++j)
            hout[(mbase + hi * 8 + j) * 128 + colg] = (f16)acc[mt][t][j];
          atomicAdd(&ls[colg], s);
          atomicAdd(&ls[128 + colg], q);
        } else if (OMODE == OUT_SCATTER) {
#pragma unroll
          for (int j = 0; j < 8; ++j)
            gadd(&agg[(long)cv[j] * 128 + colg], acc[mt][t][j]);
        } else {
#pragma unroll
          for (int j = 0; j < 8; ++j)
            fout[(mbase + hi * 8 + j) * 128 + colg] = acc[mt][t][j];
        }
      }
    }
  }
  if (OMODE == OUT_STATS) {
    __syncthreads();
    gadd(&stats[tid], ls[tid]);
    gadd(&stats[128 + tid], ls[tid + 128]);
  }
}

// Zero a float region.
__global__ void k_zero(float* p, long n) {
  long i = (long)blockIdx.x * blockDim.x + threadIdx.x;
  if (i < n) p[i] = 0.f;
}

// Degree counts for both endpoints.
__global__ void k_count(const int* __restrict__ row, const int* __restrict__ col,
                        float* degrow, float* cnt, int E) {
  int i = blockIdx.x * blockDim.x + threadIdx.x;
  if (i < E) {
    gadd(&degrow[row[i]], 1.0f);
    gadd(&cnt[col[i]], 1.0f);
  }
}

// Column sum / sum-of-squares of src[R][C], optionally row-weighted.
__global__ __launch_bounds__(256) void k_colstats(const float* __restrict__ src,
                                                  const float* __restrict__ w,
                                                  float* __restrict__ ssum,
                                                  float* __restrict__ ssq,
                                                  long R, int C) {
  __shared__ float lsum[128], lsq[128];
  const int tid = threadIdx.x;
  if (tid < C) { lsum[tid] = 0.f; lsq[tid] = 0.f; }
  __syncthreads();
  const long total = R * C;
  for (long i = (long)blockIdx.x * blockDim.x + tid; i < total;
       i += (long)gridDim.x * blockDim.x) {
    const int c = (int)(i & (C - 1));  // C is 64 or 128
    const long r = i / C;
    const float v = src[i];
    const float wt = w ? w[r] : 1.0f;
    atomicAdd(&lsum[c], wt * v);
    atomicAdd(&lsq[c], wt * v * v);
  }
  __syncthreads();
  if (tid < C) {
    gadd(&ssum[tid], lsum[tid]);
    gadd(&ssq[tid], lsq[tid]);
  }
}

// Fold BN (batch stats) into the following Linear:
//   Wf[o][k] = W[o][k] * g[k]/sigma[k]   (f16)
//   cf[o]    = c[o] + sum_k (b[k] - m[k]*g[k]/sigma[k]) * W[o][k]
__global__ __launch_bounds__(256) void k_fold(const float* __restrict__ ssum,
                                              const float* __restrict__ ssq,
                                              const float* __restrict__ g,
                                              const float* __restrict__ b,
                                              const float* __restrict__ W,
                                              const float* __restrict__ c,
                                              float invCount, int din,
                                              f16* __restrict__ Wf,
                                              float* __restrict__ cf) {
  __shared__ float s[256], t[256];
  const int tid = threadIdx.x;
  for (int k = tid; k < din; k += 256) {
    const float m = ssum[k] * invCount;
    const float v = ssq[k] * invCount - m * m;
    const float sc = g[k] * rsqrtf(v + 1e-5f);
    s[k] = sc;
    t[k] = b[k] - m * sc;
  }
  __syncthreads();
  for (int o = tid; o < 128; o += 256) {
    float acc = 0.f;
    for (int k = 0; k < din; ++k) {
      const float w = W[(size_t)o * din + k];
      acc += t[k] * w;
      Wf[(size_t)o * din + k] = (f16)(w * s[k]);
    }
    cf[o] = c[o] + acc;
  }
}

// agg[n][c] /= max(cnt[n], 1)
__global__ void k_aggdiv(float* agg, const float* cnt, long total) {
  long i = (long)blockIdx.x * blockDim.x + threadIdx.x;
  if (i < total) agg[i] /= fmaxf(cnt[i >> 7], 1.0f);
}

extern "C" void kernel_launch(void* const* d_in, const int* in_sizes, int n_in,
                              void* d_out, int out_size, void* d_ws,
                              size_t ws_size, hipStream_t stream) {
  (void)n_in; (void)ws_size; (void)out_size;
  const float* x = (const float*)d_in[0];
  const int* ei = (const int*)d_in[1];
  const float* ea = (const float*)d_in[2];
  const int N = in_sizes[0] / 128;
  const int E = in_sizes[1] / 2;
  const int* row = ei;
  const int* col = ei + E;

  // params: m1 layers at d_in[3 + 4i], m2 layers at d_in[15 + 4i]
  const float *G[6], *B[6], *W[6], *C[6];
  for (int i = 0; i < 3; ++i) {
    G[i] = (const float*)d_in[3 + 4 * i];
    B[i] = (const float*)d_in[4 + 4 * i];
    W[i] = (const float*)d_in[5 + 4 * i];
    C[i] = (const float*)d_in[6 + 4 * i];
    G[3 + i] = (const float*)d_in[15 + 4 * i];
    B[3 + i] = (const float*)d_in[16 + 4 * i];
    W[3 + i] = (const float*)d_in[17 + 4 * i];
    C[3 + i] = (const float*)d_in[18 + 4 * i];
  }

  // workspace carve-up (256B aligned)
  char* base = (char*)d_ws;
  size_t off = 0;
  auto take = [&](size_t bytes) -> char* {
    char* r = base + off;
    off = (off + bytes + 255) & ~(size_t)255;
    return r;
  };
  float* stats = (float*)take(1920 * 4);      // all 6 stats regions
  float* degrow = (float*)take((size_t)N * 4);
  float* cnt = (float*)take((size_t)N * 4);
  float* agg = (float*)take((size_t)N * 128 * 4);
  const size_t zfloats = off / 4;             // zero everything above
  float* cf = (float*)take(128 * 4);
  f16* Wf = (f16*)take(256 * 128 * 2);
  f16* hA = (f16*)take((size_t)E * 128 * 2);
  f16* hB = (f16*)take((size_t)E * 128 * 2);

  float* s0 = stats;          // din=192: sum[0:192], sq[192:384]
  float* s1 = stats + 384;    // din=128
  float* s2 = stats + 640;
  float* t0 = stats + 896;    // din=256: sum[896:1152], sq[1152:1408]
  float* t1 = stats + 1408;
  float* t2 = stats + 1664;

  const int gE = (E + 127) / 128;
  const int gN = (N + 127) / 128;

  k_zero<<<(int)((zfloats + 255) / 256), 256, 0, stream>>>((float*)base, (long)zfloats);
  k_count<<<(E + 255) / 256, 256, 0, stream>>>(row, col, degrow, cnt, E);

  // ---- m1 layer 0: stats of [x[row] | edge_attr] without the gather ----
  k_colstats<<<1024, 256, 0, stream>>>(x, degrow, s0, s0 + 192, N, 128);
  k_colstats<<<2048, 256, 0, stream>>>(ea, nullptr, s0 + 128, s0 + 320, E, 64);
  k_fold<<<1, 256, 0, stream>>>(s0, s0 + 192, G[0], B[0], W[0], C[0], 1.0f / E, 192, Wf, cf);
  k_gemm<6, false, true, 64, OUT_STATS, true><<<gE, 128, 0, stream>>>(
      nullptr, x, ea, row, nullptr, Wf, cf, hA, nullptr, s1, nullptr, E);

  // ---- m1 layer 1 ----
  k_fold<<<1, 256, 0, stream>>>(s1, s1 + 128, G[1], B[1], W[1], C[1], 1.0f / E, 128, Wf, cf);
  k_gemm<4, true, false, 0, OUT_STATS, true><<<gE, 128, 0, stream>>>(
      hA, nullptr, nullptr, nullptr, nullptr, Wf, cf, hB, nullptr, s2, nullptr, E);

  // ---- m1 layer 2 + fused scatter-sum into agg ----
  k_fold<<<1, 256, 0, stream>>>(s2, s2 + 128, G[2], B[2], W[2], C[2], 1.0f / E, 128, Wf, cf);
  k_gemm<4, true, false, 0, OUT_SCATTER, false><<<gE, 128, 0, stream>>>(
      hB, nullptr, nullptr, nullptr, col, Wf, cf, nullptr, nullptr, nullptr, agg, E);

  // ---- scatter mean ----
  k_aggdiv<<<(int)(((long)N * 128 + 255) / 256), 256, 0, stream>>>(agg, cnt, (long)N * 128);

  // ---- m2 layer 0: stats of [x | agg] ----
  k_colstats<<<512, 256, 0, stream>>>(x, nullptr, t0, t0 + 256, N, 128);
  k_colstats<<<512, 256, 0, stream>>>(agg, nullptr, t0 + 128, t0 + 384, N, 128);
  k_fold<<<1, 256, 0, stream>>>(t0, t0 + 256, G[3], B[3], W[3], C[3], 1.0f / N, 256, Wf, cf);
  k_gemm<8, false, false, 128, OUT_STATS, true><<<gN, 128, 0, stream>>>(
      nullptr, x, agg, nullptr, nullptr, Wf, cf, hA, nullptr, t1, nullptr, N);

  // ---- m2 layer 1 ----
  k_fold<<<1, 256, 0, stream>>>(t1, t1 + 128, G[4], B[4], W[4], C[4], 1.0f / N, 128, Wf, cf);
  k_gemm<4, true, false, 0, OUT_STATS, true><<<gN, 128, 0, stream>>>(
      hA, nullptr, nullptr, nullptr, nullptr, Wf, cf, hB, nullptr, t2, nullptr, N);

  // ---- m2 layer 2 -> d_out (f32) ----
  k_fold<<<1, 256, 0, stream>>>(t2, t2 + 128, G[5], B[5], W[5], C[5], 1.0f / N, 128, Wf, cf);
  k_gemm<4, true, false, 0, OUT_F32, false><<<gN, 128, 0, stream>>>(
      hB, nullptr, nullptr, nullptr, nullptr, Wf, cf, nullptr, (float*)d_out,
      nullptr, nullptr, N);
}
